// DINAttention_2439541424840
// MI455X (gfx1250) — matrix operand order repaired and checked by reading, not compiled, not run on hardware
//
#include <hip/hip_runtime.h>
#include <cstdint>

#define NEGV (-10000.0f)

typedef __bf16 v8bf  __attribute__((ext_vector_type(8)));
typedef __bf16 v16bf __attribute__((ext_vector_type(16)));
typedef float  v8f   __attribute__((ext_vector_type(8)));

// ---------------- problem constants (match reference) ----------------
constexpr int D  = 128;   // key/query dim
constexpr int L  = 200;   // history length
constexpr int H1 = 256;   // hidden 1
constexpr int H2 = 128;   // hidden 2
constexpr int NT = 13;    // token tiles of 16 (208 padded)
constexpr int LP = NT * 16;

// ---------------- LDS layout (bytes) ----------------
constexpr int WSTRIDE  = 136;                    // weff row stride (bf16 elems)
constexpr int HSTRIDE  = 264;                    // h1s row stride (bf16 elems, 256+8 pad)
constexpr int OFF_WEFF = 0;                      // 256 x 136 bf16 = 69632
constexpr int OFF_BEFF = 69632;                  // 256 f32        = 1024
constexpr int OFF_H1S  = 70656;                  // 8 waves x 16 x 264 bf16 = 67584
constexpr int OFF_B2L  = 138240;                 // 128 f32
constexpr int OFF_W3L  = 138752;                 // 128 f32
constexpr int OFF_SC   = 139264;                 // 208 f32 = 832
constexpr int OFF_RED  = 140096;                 // 256 f32 = 1024
constexpr int SMEM_BYTES = 141120;               // ~138 KB of the WGP's 320 KB LDS

__device__ __forceinline__ v16bf cat16(v8bf a, v8bf b) {
  return __builtin_shufflevector(a, b, 0,1,2,3,4,5,6,7,8,9,10,11,12,13,14,15);
}
__device__ __forceinline__ v8bf cvt8(float4 a, float4 b) {
  v8bf r;
  r[0]=(__bf16)a.x; r[1]=(__bf16)a.y; r[2]=(__bf16)a.z; r[3]=(__bf16)a.w;
  r[4]=(__bf16)b.x; r[5]=(__bf16)b.y; r[6]=(__bf16)b.z; r[7]=(__bf16)b.w;
  return r;
}
__device__ __forceinline__ v8f wmma_bf16(v16bf a, v16bf b, v8f c) {
  // D = A(16x32 bf16) * B(32x16 bf16) + C(16x16 f32)
  return __builtin_amdgcn_wmma_f32_16x16x32_bf16(false, a, false, b, (short)0, c, false, false);
}
__device__ __forceinline__ v8f zero8() {
  v8f z = {0.f,0.f,0.f,0.f,0.f,0.f,0.f,0.f};
  return z;
}

// Convert W2 (f32 [128][256], row-major g-major) to bf16 in workspace.
__global__ void cvt_w2_kernel(const float* __restrict__ W2, __bf16* __restrict__ outw) {
  int i = blockIdx.x * blockDim.x + threadIdx.x;
  outw[i] = (__bf16)W2[i];
}

__global__ __launch_bounds__(256) void din_att_kernel(
    const float* __restrict__ query, const float* __restrict__ keys,
    const int* __restrict__ mask,    const float* __restrict__ W1,
    const float* __restrict__ b1,    const float* __restrict__ pa1,
    const float* __restrict__ b2,    const float* __restrict__ pa2,
    const float* __restrict__ W3,    const float* __restrict__ pb3,
    const __bf16* __restrict__ W2bf, float* __restrict__ out)
{
  extern __shared__ char smem[];
  __bf16* weff = (__bf16*)(smem + OFF_WEFF);   // [n=0..255][k=0..127], stride 136
  float*  beff = (float*)(smem + OFF_BEFF);    // [256]
  __bf16* h1s  = (__bf16*)(smem + OFF_H1S);    // per-wave [16][256], stride 264
  float*  b2l  = (float*)(smem + OFF_B2L);     // [128]
  float*  w3l  = (float*)(smem + OFF_W3L);     // [128]
  float*  sc   = (float*)(smem + OFF_SC);      // [208] scores -> weights
  float*  red  = (float*)(smem + OFF_RED);     // [256] reduction scratch

  const int b     = blockIdx.x;
  const int tid   = threadIdx.x;
  const int wave  = tid >> 5;
  const int lane  = tid & 31;
  const int ln16  = lane & 15;
  const int hi    = lane >> 4;              // 0: lanes 0-15, 1: lanes 16-31
  const int wbase = wave * 16 * HSTRIDE;    // bf16 elements

  const float a1v = pa1[0], a2v = pa2[0], b3v = pb3[0];
  const float* qr   = query + (size_t)b * D;
  const int*   mrow = mask  + (size_t)b * L;
  const float* keyb = keys  + (size_t)b * L * D;

  // ---- prologue: fold query into effective layer-1 weights/bias ----
  // h1_pre[n] = beff[n] + sum_f Weff[n][f] * k[f]
  // beff[n]   = b1[n] + sum_f (W1q[n,f] + W1d[n,f]) * q[f]
  // Weff[n][f]= W1k[n,f] - W1d[n,f] + W1p[n,f]*q[f]
  {
    const float* w1r = W1 + (size_t)tid * (4 * D);
    float acc = b1[tid];
    #pragma unroll 4
    for (int f = 0; f < D; ++f) {
      float qf = qr[f];
      acc += (w1r[f] + w1r[2*D + f]) * qf;
      weff[tid*WSTRIDE + f] = (__bf16)(w1r[D + f] - w1r[2*D + f] + w1r[3*D + f] * qf);
    }
    beff[tid] = acc;
    if (tid < H2) { b2l[tid] = b2[tid]; w3l[tid] = W3[tid]; }
  }
  __syncthreads();

  // ---- per-wave 16-token tiles through MLP + scores ----
  for (int tile = wave; tile < NT; tile += 8) {
    const int t = tile * 16 + ln16;           // token of this lane (A-matrix M)
    // Pad tokens (t >= L) clamp to row L-1; their scores are overwritten with
    // NEG below, and each WMMA output row depends only on its own token row,
    // so no zeroing of the A fragment is needed.
    const float* krow = keyb + (size_t)(t < L ? t : L - 1) * D;

    // prefetch this wave's next tile of keys (global_prefetch_b8)
    int tn = t + 128;
    if (tn < L) __builtin_prefetch(keyb + (size_t)tn * D, 0, 1);

    // Issue all 16 key loads into distinct temporaries (one wait, not 16),
    // then convert. ISA 16-bit A layout:
    //   elems 0..7 -> K = k*32 + 8*hi + (0..7); elems 8..15 -> +16
    float4 kf[16];
    #pragma unroll
    for (int k = 0; k < 4; ++k) {
      const float* p0 = krow + k*32 + 8*hi;
      kf[4*k+0] = *(const float4*)(p0);
      kf[4*k+1] = *(const float4*)(p0 + 4);
      kf[4*k+2] = *(const float4*)(p0 + 16);
      kf[4*k+3] = *(const float4*)(p0 + 20);
    }
    v16bf akey[4];
    #pragma unroll
    for (int k = 0; k < 4; ++k)
      akey[k] = cat16(cvt8(kf[4*k+0], kf[4*k+1]), cvt8(kf[4*k+2], kf[4*k+3]));

    // ---- layer 1 (K=128, N=256 in two halves); acc1 reused per half ----
    #pragma unroll
    for (int half = 0; half < 2; ++half) {
      v8f acc1[8];
      #pragma unroll
      for (int n = 0; n < 8; ++n) acc1[n] = zero8();
      #pragma unroll
      for (int k = 0; k < 4; ++k) {
        #pragma unroll
        for (int n = 0; n < 8; ++n) {
          // B layout: N = lane%16 (row n of Weff), elems 0..7 -> K = k*32 + 8*hi + e; elems 8..15 -> +16
          const __bf16* bp = weff + ((half*8 + n)*16 + ln16)*WSTRIDE + k*32 + hi*8;
          v16bf bf = cat16(*(const v8bf*)bp, *(const v8bf*)(bp + 16));
          acc1[n] = wmma_bf16(akey[k], bf, acc1[n]);
        }
      }
      // bias + PReLU in D layout (M = r + 8*hi, N = ln16); stage bf16 to LDS
      #pragma unroll
      for (int n = 0; n < 8; ++n) {
        float bb = beff[(half*8 + n)*16 + ln16];
        #pragma unroll
        for (int r = 0; r < 8; ++r) {
          float v = acc1[n][r] + bb;
          v = (v >= 0.0f) ? v : a1v * v;
          h1s[wbase + (r + 8*hi)*HSTRIDE + half*128 + n*16 + ln16] = (__bf16)v;
        }
      }
    }
    asm volatile("s_wait_dscnt 0x0" ::: "memory"); // cross-lane LDS store->load in-wave

    // ---- layer 2 (K=256); acc1 is dead here, so acc2 has the registers ----
    v8f acc2[8];
    #pragma unroll
    for (int n = 0; n < 8; ++n) acc2[n] = zero8();
    #pragma unroll
    for (int k2 = 0; k2 < 8; ++k2) {
      const __bf16* ap = h1s + wbase + ln16*HSTRIDE + k2*32 + 8*hi;
      v16bf af = cat16(*(const v8bf*)ap, *(const v8bf*)(ap + 16));
      #pragma unroll
      for (int n2 = 0; n2 < 8; ++n2) {
        const __bf16* wp = W2bf + (size_t)(n2*16 + ln16)*H1 + k2*32 + hi*8;
        v16bf bf = cat16(*(const v8bf*)wp, *(const v8bf*)(wp + 16));
        acc2[n2] = wmma_bf16(af, bf, acc2[n2]);
      }
    }

    // ---- layer-2 epilogue fused with layer-3 dot (W3) ----
    float p[8];
    #pragma unroll
    for (int r = 0; r < 8; ++r) p[r] = 0.0f;
    #pragma unroll
    for (int n2 = 0; n2 < 8; ++n2) {
      int g = n2*16 + ln16;
      float bb = b2l[g], wg = w3l[g];
      #pragma unroll
      for (int r = 0; r < 8; ++r) {
        float v = acc2[n2][r] + bb;
        v = (v >= 0.0f) ? v : a2v * v;
        p[r] += v * wg;
      }
    }
    // reduce partials across the 16 lanes of each half (wave32 shuffles)
    #pragma unroll
    for (int off = 1; off < 16; off <<= 1) {
      #pragma unroll
      for (int r = 0; r < 8; ++r) p[r] += __shfl_xor(p[r], off, 32);
    }
    if (ln16 == 0) {  // lanes 0 (tokens 0..7) and 16 (tokens 8..15)
      #pragma unroll
      for (int r = 0; r < 8; ++r) {
        int tt = tile*16 + 8*hi + r;
        int ok = (tt < L) ? mrow[tt] : 0;
        sc[tt] = ok ? (p[r] + b3v) : NEGV;
      }
    }
  }
  __syncthreads();

  // ---- softmax over L (NEG-masked like reference), then zero masked ----
  float s = (tid < LP) ? sc[tid] : NEGV;
  red[tid] = s; __syncthreads();
  for (int off = 128; off > 0; off >>= 1) {
    if (tid < off) red[tid] = fmaxf(red[tid], red[tid + off]);
    __syncthreads();
  }
  float mx = red[0]; __syncthreads();
  float e = (tid < LP) ? __expf(s - mx) : 0.0f;
  red[tid] = e; __syncthreads();
  for (int off = 128; off > 0; off >>= 1) {
    if (tid < off) red[tid] += red[tid + off];
    __syncthreads();
  }
  float inv = 1.0f / red[0]; __syncthreads();
  if (tid < LP) {
    int ok = (tid < L) ? mrow[tid] : 0;
    sc[tid] = ok ? e * inv : 0.0f;
  }
  __syncthreads();

  // ---- out[b][d] = sum_t w[t] * keys[b][t][d] ----
  int d = tid & 127, hh = tid >> 7;
  float acc = 0.0f;
  for (int t2 = hh; t2 < L; t2 += 2) acc += sc[t2] * keyb[(size_t)t2 * D + d];
  red[tid] = acc; __syncthreads();
  if (tid < D) out[(size_t)b * D + tid] = red[tid] + red[tid + 128];
}

extern "C" void kernel_launch(void* const* d_in, const int* in_sizes, int n_in,
                              void* d_out, int out_size, void* d_ws, size_t ws_size,
                              hipStream_t stream) {
  const float* query = (const float*)d_in[0];
  const float* keys  = (const float*)d_in[1];
  const int*   mask  = (const int*)d_in[2];
  const float* W1    = (const float*)d_in[3];
  const float* b1    = (const float*)d_in[4];
  const float* a1    = (const float*)d_in[5];
  const float* W2    = (const float*)d_in[6];
  const float* b2    = (const float*)d_in[7];
  const float* a2    = (const float*)d_in[8];
  const float* W3    = (const float*)d_in[9];
  const float* b3    = (const float*)d_in[10];
  float*  out  = (float*)d_out;
  __bf16* W2bf = (__bf16*)d_ws;   // 128*256 bf16 = 64 KB of workspace

  const int B = in_sizes[0] / D;  // 2048

  cvt_w2_kernel<<<(H1 * H2) / 256, 256, 0, stream>>>(W2, W2bf);

  hipFuncSetAttribute(reinterpret_cast<const void*>(din_att_kernel),
                      hipFuncAttributeMaxDynamicSharedMemorySize, SMEM_BYTES);
  din_att_kernel<<<B, 256, SMEM_BYTES, stream>>>(query, keys, mask, W1, b1, a1,
                                                 b2, a2, W3, b3, W2bf, out);
}
